// PointerNetwork_34935263985847
// MI455X (gfx1250) — matrix operand order repaired and checked by reading, not compiled, not run on hardware
//
#include <hip/hip_runtime.h>
#include <hip/hip_bf16.h>
#include <math.h>

// Problem constants (from reference)
#define H   128
#define S   64
#define B   256
#define TB  16          // batch rows per block (= WMMA M)
#define LDX 132         // padded LDS row stride (floats) for s_uh

typedef __attribute__((ext_vector_type(16))) __bf16 v16bf;
typedef __attribute__((ext_vector_type(8)))  float  v8f;
typedef __attribute__((ext_vector_type(4)))  unsigned u32x4;
typedef __attribute__((ext_vector_type(8)))  int      i32x8;
typedef __attribute__((ext_vector_type(4)))  int      i32x4;

union BFrag {
  unsigned short us[16];
  unsigned       u32[8];
  v16bf          v;
};

__device__ __forceinline__ unsigned short f2bf(float f) {
  unsigned u = __float_as_uint(f);
  unsigned r = u + 0x7fffu + ((u >> 16) & 1u);   // RNE
  return (unsigned short)(r >> 16);
}

__device__ __forceinline__ v8f wmma_bf16(v16bf a, v16bf b, v8f c) {
  return __builtin_amdgcn_wmma_f32_16x16x32_bf16(false, a, false, b, (short)0, c,
                                                 false, false);
}

// Fragment (A or B, 16 bf16 elems/lane) from packed storage: 8 contiguous dwords
// per lane per tile. Works for LDS or global base pointers.
__device__ __forceinline__ v16bf frag_packed(const unsigned* base, int tile) {
  int lane = threadIdx.x & 31;
  const unsigned* p = base + ((tile << 5) + lane) * 8;
  BFrag f;
#pragma unroll
  for (int v = 0; v < 8; ++v) f.u32[v] = p[v];
  return f.v;
}

// Scatter one bf16 element (m,k) of a 16x128 matrix into A-fragment-packed LDS.
// A layout (ISA 7.12.2, 16-bit A 16x32): lanes0-15 hold K {0..7,16..23},
// lanes16-31 hold K {8..15,24..31}.
__device__ __forceinline__ void pack_store16(unsigned short* us_base, int m,
                                             int k, unsigned short val) {
  int kt = k >> 5, kl = k & 31;
  int lane = m + (((kl >> 3) & 1) << 4);
  int v = ((kl & 16) >> 2) + ((kl & 7) >> 1);
  us_base[((((kt << 5) + lane) << 3) + v) * 2 + (kl & 1)] = val;
}

__device__ __forceinline__ unsigned rotl32(unsigned x, int d) {
  return (x << d) | (x >> (32 - d));
}

// JAX threefry2x32 (20 rounds)
__device__ __forceinline__ void threefry2x32(unsigned k0, unsigned k1,
                                             unsigned x0, unsigned x1,
                                             unsigned* o0, unsigned* o1) {
  unsigned k2 = k0 ^ k1 ^ 0x1BD11BDAu;
  x0 += k0; x1 += k1;
#define TF_R(r) { x0 += x1; x1 = rotl32(x1, r); x1 ^= x0; }
  TF_R(13) TF_R(15) TF_R(26) TF_R(6)  x0 += k1; x1 += k2 + 1u;
  TF_R(17) TF_R(29) TF_R(16) TF_R(24) x0 += k2; x1 += k0 + 2u;
  TF_R(13) TF_R(15) TF_R(26) TF_R(6)  x0 += k0; x1 += k1 + 3u;
  TF_R(17) TF_R(29) TF_R(16) TF_R(24) x0 += k1; x1 += k2 + 4u;
  TF_R(13) TF_R(15) TF_R(26) TF_R(6)  x0 += k2; x1 += k0 + 5u;
#undef TF_R
  *o0 = x0; *o1 = x1;
}

__device__ __forceinline__ float fast_tanh(float x) {
  float e = __expf(2.0f * x);                       // inf for big x -> 1
  return 1.0f - 2.0f * __builtin_amdgcn_rcpf(e + 1.0f);
}
__device__ __forceinline__ float fast_sigmoid(float x) {
  return __builtin_amdgcn_rcpf(1.0f + __expf(-x));
}

// Issue one TDM 1-row copy (n8 * 8 bytes) global -> LDS. D# per ISA ch8.
__device__ __forceinline__ void tdm_load_lds(unsigned lds_addr,
                                             const void* gptr, unsigned n8) {
  unsigned long long ga = (unsigned long long)(size_t)gptr;
  u32x4 g0;
  g0[0] = 1u;                                        // count=1 (valid), user mode
  g0[1] = lds_addr;                                  // LDS byte address
  g0[2] = (unsigned)(ga & 0xffffffffu);              // global_addr lo
  g0[3] = (unsigned)((ga >> 32) & 0x1ffffffu) | (2u << 30);  // addr hi | type=2
  i32x8 g1 = {0, 0, 0, 0, 0, 0, 0, 0};
  g1[0] = (3 << 16);                                 // data_size = 8 bytes
  g1[1] = (int)((n8 & 0xffffu) << 16);               // tensor_dim0 lo16
  g1[2] = (int)(((n8 >> 16) & 0xffffu) | (1u << 16)); // tensor_dim0 hi16 | tensor_dim1=1
  g1[3] = (int)((n8 & 0xffffu) << 16);               // tile_dim0
  g1[4] = 1;                                         // tile_dim1 = 1
  g1[5] = (int)n8;                                   // tensor_dim0_stride
  i32x4 z = {0, 0, 0, 0};
#if defined(__clang_major__) && (__clang_major__ >= 23)
  i32x8 z8 = {0, 0, 0, 0, 0, 0, 0, 0};
  __builtin_amdgcn_tensor_load_to_lds(g0, g1, z, z, z8, 0);
#else
  __builtin_amdgcn_tensor_load_to_lds(g0, g1, z, z, 0);
#endif
}

// ---------------------------------------------------------------------------
// Prep: pack W^T (B-matrix, B[k][n] = W[n][k]) into WMMA B-fragment order, bf16.
__global__ void pack_b_kernel(const float* __restrict__ W, int ldw, int col0,
                              int N, int K, unsigned* __restrict__ out) {
  int total = (N >> 4) * (K >> 5) * 256;
  for (int idx = blockIdx.x * blockDim.x + threadIdx.x; idx < total;
       idx += gridDim.x * blockDim.x) {
    int v    = idx & 7;
    int lane = (idx >> 3) & 31;
    int tile = idx >> 8;
    int KT = K >> 5;
    int nt = tile / KT, kt = tile - nt * KT;
    int n  = (nt << 4) + (lane & 15);
    int k  = (kt << 5) + ((lane >> 4) << 3) + ((v < 4) ? 2 * v : 16 + 2 * (v - 4));
    unsigned lo = f2bf(W[n * ldw + col0 + k]);
    unsigned hi = f2bf(W[n * ldw + col0 + k + 1]);
    out[idx] = lo | (hi << 16);
  }
}

// Prep: A2 = W_att_dyn @ dyn_w (H x 2); cdb = W_att_dyn @ dyn_b (H).
__global__ void prep_small(const float* __restrict__ W_att,
                           const float* __restrict__ dyn_w,
                           const float* __restrict__ dyn_b,
                           float* __restrict__ A2, float* __restrict__ cdb) {
  int h = blockIdx.x * blockDim.x + threadIdx.x;
  if (h >= H) return;
  float a0 = 0.f, a1 = 0.f, c = 0.f;
  for (int k = 0; k < H; ++k) {
    float w = W_att[h * (3 * H) + H + k];
    a0 += w * dyn_w[k * 2 + 0];
    a1 += w * dyn_w[k * 2 + 1];
    c  += w * dyn_b[k];
  }
  A2[h * 2] = a0; A2[h * 2 + 1] = a1; cdb[h] = c;
}

// Precompute pre[b][h][s] = (W_att[:, :H] @ static_emb[b])[h][s] + cdb[h].
__global__ __launch_bounds__(128) void pre_gemm_kernel(
    const float* __restrict__ W_att, const float* __restrict__ se,
    const float* __restrict__ cdb, float* __restrict__ pre) {
  int blk  = blockIdx.x;
  int b    = blk >> 3, mt = blk & 7;
  int wave = threadIdx.x >> 5, lane = threadIdx.x & 31;
  int s0   = wave * 16;
  v8f acc = {0.f, 0.f, 0.f, 0.f, 0.f, 0.f, 0.f, 0.f};
#pragma unroll
  for (int kt = 0; kt < 4; ++kt) {
    BFrag fa, fb;
    int mrow = mt * 16 + (lane & 15);
    int kb   = (lane >> 4) << 3;
#pragma unroll
    for (int v = 0; v < 8; ++v) {
      int k = kt * 32 + ((v < 4) ? (kb + 2 * v) : (kb + 16 + 2 * (v - 4)));
      fa.us[2 * v]     = f2bf(W_att[mrow * (3 * H) + k]);
      fa.us[2 * v + 1] = f2bf(W_att[mrow * (3 * H) + k + 1]);
    }
    int sc  = s0 + (lane & 15);
    int kb2 = (lane >> 4) << 3;
#pragma unroll
    for (int v = 0; v < 8; ++v) {
      int k = kt * 32 + ((v < 4) ? (kb2 + 2 * v) : (kb2 + 16 + 2 * (v - 4)));
      fb.us[2 * v]     = f2bf(se[(b * H + k) * S + sc]);
      fb.us[2 * v + 1] = f2bf(se[(b * H + k + 1) * S + sc]);
    }
    acc = wmma_bf16(fa.v, fb.v, acc);
  }
  int n = lane & 15, hi = lane >> 4;
#pragma unroll
  for (int r = 0; r < 8; ++r) {
    int row = mt * 16 + r + hi * 8;
    pre[(b * H + row) * S + s0 + n] = acc[r] + cdb[row];
  }
}

// ---------------------------------------------------------------------------
// Main persistent kernel: one block per 16 batch rows, sequential over 64 steps.
__global__ __launch_bounds__(256) void pointer_main(
    const float* __restrict__ se, const float* __restrict__ dynamic,
    const float* __restrict__ in_b, const float* __restrict__ b_ih,
    const float* __restrict__ b_hh, const float* __restrict__ v_att,
    const unsigned* __restrict__ pk_inw, const unsigned* __restrict__ pk_wih,
    const unsigned* __restrict__ pk_whh, const unsigned* __restrict__ pk_whT,
    const float* __restrict__ A2g, const float* __restrict__ pre,
    int* __restrict__ tours, float* __restrict__ logps) {
  // GRU weights resident in LDS (loaded once via TDM): 2 x 96 KB
  __shared__ unsigned s_wih[24576];
  __shared__ unsigned s_whh[24576];
  // decoder state, bf16 A-fragment-packed (ping-pong dec/x, plus h)
  __shared__ unsigned s_p0[1024], s_p1[1024], s_hp[1024];
  __shared__ float s_uh[TB * LDX];
  __shared__ float s_vis[TB * S];
  __shared__ float s_dyn[TB * 2 * S];
  __shared__ float s_scores[TB * S];
  __shared__ float s_v[H], s_A2[2 * H];
  __shared__ int   s_ptr[TB];

  const int tid  = threadIdx.x;
  const int wave = tid >> 5;
  const int lane = tid & 31;
  const int bg0  = blockIdx.x * TB;
  const int n16  = lane & 15;
  const int hif  = lane >> 4;

  // ---- TDM: pull both GRU weight blocks into LDS ----
  if (tid == 0) {
    tdm_load_lds((unsigned)(size_t)&s_wih[0], pk_wih, 24576 / 2);
    tdm_load_lds((unsigned)(size_t)&s_whh[0], pk_whh, 24576 / 2);
  }

  // ---- init state (overlaps with TDM) ----
  unsigned short* p0_16 = (unsigned short*)s_p0;
  unsigned short* hp_16 = (unsigned short*)s_hp;
  for (int t = tid; t < TB * H; t += 256) {
    int m = t >> 7, k = t & 127;
    pack_store16(p0_16, m, k, f2bf(se[(bg0 + m) * (H * S) + k * S]));
  }
  for (int t = tid; t < 1024; t += 256) s_hp[t] = 0u;
  for (int t = tid; t < TB * 2 * S; t += 256)
    s_dyn[t] = dynamic[bg0 * 2 * S + t];
  for (int t = tid; t < TB * S; t += 256) s_vis[t] = 0.f;
  for (int t = tid; t < H; t += 256) s_v[t] = v_att[t];
  for (int t = tid; t < 2 * H; t += 256) s_A2[t] = A2g[t];

  __builtin_amdgcn_s_wait_tensorcnt((short)0);
  __syncthreads();

  // step-invariant per-thread constants (wave owns columns jl = wave*16+n16)
  const int jl = wave * 16 + n16;
  const float bias_x = in_b[jl];
  const float bir = b_ih[jl],         bhr = b_hh[jl];
  const float biz = b_ih[H + jl],     bhz = b_hh[H + jl];
  const float bin = b_ih[2 * H + jl], bhn = b_hh[2 * H + jl];
  float hreg[8];                        // fp32 master copy of h (m=r+hif*8, col jl)
#pragma unroll
  for (int r = 0; r < 8; ++r) hreg[r] = 0.f;

  for (int i = 0; i < S; ++i) {
    unsigned* s_dec = (i & 1) ? s_p1 : s_p0;
    unsigned* s_xp  = (i & 1) ? s_p0 : s_p1;
    unsigned short* xp16 = (unsigned short*)s_xp;

    // ---- (a) x = dec_in @ in_w^T + in_b : wave w owns n-tile w ----
    {
      v8f acc = {0.f, 0.f, 0.f, 0.f, 0.f, 0.f, 0.f, 0.f};
#pragma unroll
      for (int kt = 0; kt < 4; ++kt)
        acc = wmma_bf16(frag_packed(s_dec, kt),
                        frag_packed(pk_inw, wave * 4 + kt), acc);
#pragma unroll
      for (int r = 0; r < 8; ++r)
        pack_store16(xp16, r + hif * 8, jl, f2bf(acc[r] + bias_x));
      __syncthreads();
    }

    // ---- (b) GRU gates: wave w owns column slice jl of all 3 gate chunks ----
    {
      v16bf ax[4], ah[4];
#pragma unroll
      for (int kt = 0; kt < 4; ++kt) {
        ax[kt] = frag_packed(s_xp, kt);
        ah[kt] = frag_packed(s_hp, kt);
      }
      v8f gi_[3], gh_[3];
#pragma unroll
      for (int c = 0; c < 3; ++c) {
        v8f gi = {0.f, 0.f, 0.f, 0.f, 0.f, 0.f, 0.f, 0.f};
        v8f gh = {0.f, 0.f, 0.f, 0.f, 0.f, 0.f, 0.f, 0.f};
        int nt = c * 8 + wave;
#pragma unroll
        for (int kt = 0; kt < 4; ++kt) {
          gi = wmma_bf16(ax[kt], frag_packed(s_wih, nt * 4 + kt), gi);
          gh = wmma_bf16(ah[kt], frag_packed(s_whh, nt * 4 + kt), gh);
        }
        gi_[c] = gi; gh_[c] = gh;
      }
      float hn[8];
#pragma unroll
      for (int r = 0; r < 8; ++r) {
        float rg = fast_sigmoid((gi_[0][r] + bir) + (gh_[0][r] + bhr));
        float zg = fast_sigmoid((gi_[1][r] + biz) + (gh_[1][r] + bhz));
        float ng = fast_tanh((gi_[2][r] + bin) + rg * (gh_[2][r] + bhn));
        hn[r] = (1.f - zg) * ng + zg * hreg[r];
      }
      __syncthreads();   // everyone done reading old s_hp
#pragma unroll
      for (int r = 0; r < 8; ++r) {
        pack_store16(hp_16, r + hif * 8, jl, f2bf(hn[r]));
        hreg[r] = hn[r];
      }
      __syncthreads();
    }

    // ---- (c) u_h = h_new @ Wh^T : wave w owns n-tile w ----
    {
      v8f acc = {0.f, 0.f, 0.f, 0.f, 0.f, 0.f, 0.f, 0.f};
#pragma unroll
      for (int kt = 0; kt < 4; ++kt)
        acc = wmma_bf16(frag_packed(s_hp, kt),
                        frag_packed(pk_whT, wave * 4 + kt), acc);
#pragma unroll
      for (int r = 0; r < 8; ++r)
        s_uh[(r + hif * 8) * LDX + jl] = acc[r];
      __syncthreads();
    }

    // ---- (d) scores[m][s] = v . tanh(pre + A2*dyn + u_h) ----
    for (int p = tid; p < TB * S; p += 256) {
      int m = p >> 6, s = p & 63;
      const float* pr = pre + (bg0 + m) * (H * S) + s;
      __builtin_prefetch(pr, 0, 1);
      float d0 = s_dyn[m * 2 * S + s], d1 = s_dyn[m * 2 * S + S + s];
      float acc = 0.f;
#pragma unroll 4
      for (int h = 0; h < H; ++h) {
        float u = pr[h * S] + s_A2[2 * h] * d0 + s_A2[2 * h + 1] * d1 +
                  s_uh[m * LDX + h];
        acc += s_v[h] * fast_tanh(u);
      }
      s_scores[p] = acc;
    }
    __syncthreads();

    // ---- (e) masked Gumbel sampling + logp ; one row per wave per pass ----
    {
      unsigned sk0, sk1;   // step key = threefry(key(42), [0, i])  (JAX fold_in)
      threefry2x32(0u, 42u, 0u, (unsigned)i, &sk0, &sk1);
      for (int rr = 0; rr < 2; ++rr) {
        int m  = wave * 2 + rr;
        int bg = bg0 + m;
        float best = -1e30f; int bidx = 0; float lmax = -1e30f;
        float lg2[2];
#pragma unroll
        for (int t = 0; t < 2; ++t) {
          int s = lane + 32 * t;
          bool msk = (i == 0) ? (s == 0) : (s_vis[m * S + s] == 0.f);
          float lg = msk ? s_scores[m * S + s] : -1e9f;
          lg2[t] = lg;
          lmax = fmaxf(lmax, lg);
          unsigned e = (unsigned)(bg * S + s), o0, o1, bits;
          if (e < (unsigned)(B * S / 2)) {
            threefry2x32(sk0, sk1, e, e + (unsigned)(B * S / 2), &o0, &o1);
            bits = o0;
          } else {
            threefry2x32(sk0, sk1, e - (unsigned)(B * S / 2), e, &o0, &o1);
            bits = o1;
          }
          float uf = __uint_as_float((bits >> 9) | 0x3f800000u) - 1.0f;
          uf = fmaxf(1.1754944e-38f,
                     uf * (1.0f - 1.1754944e-38f) + 1.1754944e-38f);
          float g = -logf(-logf(uf));
          float val = lg + g;
          if (val > best || (val == best && s < bidx)) { best = val; bidx = s; }
        }
#pragma unroll
        for (int off = 16; off > 0; off >>= 1) {
          float ob = __shfl_xor(best, off, 32);
          int   oi = __shfl_xor(bidx, off, 32);
          float om = __shfl_xor(lmax, off, 32);
          if (ob > best || (ob == best && oi < bidx)) { best = ob; bidx = oi; }
          lmax = fmaxf(lmax, om);
        }
        float sume = __expf(lg2[0] - lmax) + __expf(lg2[1] - lmax);
#pragma unroll
        for (int off = 16; off > 0; off >>= 1)
          sume += __shfl_xor(sume, off, 32);
        if (lane == 0) {
          bool mp = (i == 0) ? (bidx == 0) : (s_vis[m * S + bidx] == 0.f);
          float lgp = (mp ? s_scores[m * S + bidx] : -1e9f) - lmax - logf(sume);
          s_ptr[m] = bidx;
          tours[bg * S + i] = bidx;
          logps[bg * S + i] = lgp;
        }
      }
    }
    __syncthreads();

    // ---- (f) state updates: visited, dyn *= (1-sel) (dec_in=x via ping-pong) --
    if (tid < TB) {
      int pz = s_ptr[tid];
      s_vis[tid * S + pz] = 1.0f;
      s_dyn[tid * 2 * S + pz] = 0.f;
      s_dyn[tid * 2 * S + S + pz] = 0.f;
    }
    __syncthreads();
  }
}

// ---------------------------------------------------------------------------
extern "C" void kernel_launch(void* const* d_in, const int* in_sizes, int n_in,
                              void* d_out, int out_size, void* d_ws, size_t ws_size,
                              hipStream_t stream) {
  (void)in_sizes; (void)n_in; (void)out_size; (void)ws_size;
  const float* se      = (const float*)d_in[1];
  const float* dynamic = (const float*)d_in[2];
  const float* dyn_w   = (const float*)d_in[3];
  const float* dyn_b   = (const float*)d_in[4];
  const float* in_w    = (const float*)d_in[5];
  const float* in_b    = (const float*)d_in[6];
  const float* w_ih    = (const float*)d_in[7];
  const float* w_hh    = (const float*)d_in[8];
  const float* b_ih    = (const float*)d_in[9];
  const float* b_hh    = (const float*)d_in[10];
  const float* W_att   = (const float*)d_in[11];
  const float* v_att   = (const float*)d_in[12];

  // workspace layout (dword units)
  unsigned* wsu = (unsigned*)d_ws;
  float*    wsf = (float*)d_ws;
  const size_t O_INW = 0;          // 8192
  const size_t O_WIH = 8192;       // 24576
  const size_t O_WHH = 32768;      // 24576
  const size_t O_WHT = 57344;      // 8192
  const size_t O_A2  = 65536;      // 256
  const size_t O_CDB = 65792;      // 128
  const size_t O_PRE = 66048;      // 2097152

  pack_b_kernel<<<32, 256, 0, stream>>>(in_w, H, 0, H, H, wsu + O_INW);
  pack_b_kernel<<<96, 256, 0, stream>>>(w_ih, H, 0, 3 * H, H, wsu + O_WIH);
  pack_b_kernel<<<96, 256, 0, stream>>>(w_hh, H, 0, 3 * H, H, wsu + O_WHH);
  pack_b_kernel<<<32, 256, 0, stream>>>(W_att, 3 * H, 2 * H, H, H, wsu + O_WHT);
  prep_small<<<1, 128, 0, stream>>>(W_att, dyn_w, dyn_b, wsf + O_A2, wsf + O_CDB);
  pre_gemm_kernel<<<B * (H / 16), 128, 0, stream>>>(W_att, se, wsf + O_CDB,
                                                    wsf + O_PRE);

  int*   tours = (int*)d_out;
  float* lps   = (float*)d_out + B * S;
  pointer_main<<<B / TB, 256, 0, stream>>>(
      se, dynamic, in_b, b_ih, b_hh, v_att,
      wsu + O_INW, wsu + O_WIH, wsu + O_WHH, wsu + O_WHT,
      wsf + O_A2, wsf + O_PRE, tours, lps);
}
